// AxialAttention3D_36636071035099
// MI455X (gfx1250) — matrix operand order repaired and checked by reading, not compiled, not run on hardware
//
#include <hip/hip_runtime.h>

typedef __attribute__((ext_vector_type(2)))  float    v2f;
typedef __attribute__((ext_vector_type(8)))  float    v8f;
typedef __attribute__((ext_vector_type(8)))  _Float16 v8h;
typedef __attribute__((ext_vector_type(16))) _Float16 v16h;

#define NTOK 4096   // D*H*W = 16^3
#define CIN  128

__device__ __forceinline__ v8f wmma_f32_k4(v2f a, v2f b, v8f c) {
  // D = A(16x4) * B(4x16) + C, fp32 (projection GEMMs, full precision)
  return __builtin_amdgcn_wmma_f32_16x16x4_f32(false, a, false, b, (short)0, c,
                                               false, false);
}

__device__ __forceinline__ v8f wmma_f16_k32(v16h a, v16h b, v8f c) {
  // D = A(16x32) * B(32x16) + C, f16 in / f32 out (attention core)
  return __builtin_amdgcn_wmma_f32_16x16x32_f16(false, a, false, b, (short)0, c,
                                                false, false);
}

// ---------------------------------------------------------------------------
// Kernel 1: qkv[br] = Wqkv_br (384x128) * x (128x4096) + b  (3 branches)
// one 16x16 tile per wave; tiles = 3 * 24 * 256 = 18432  (fp32 wmma)
// ---------------------------------------------------------------------------
__global__ __launch_bounds__(256) void qkv_proj_kernel(
    const float* __restrict__ x,
    const float* __restrict__ w0, const float* __restrict__ b0,
    const float* __restrict__ w1, const float* __restrict__ b1,
    const float* __restrict__ w2, const float* __restrict__ b2,
    float* __restrict__ qkv) {
  const int wave = threadIdx.x >> 5;
  const int lane = threadIdx.x & 31;
  const int lrow = lane & 15;
  const int hi   = lane >> 4;
  const int ka   = hi * 2;
  const int tile = blockIdx.x * 8 + wave;
  const int br   = tile / (24 * 256);
  const int rem  = tile % (24 * 256);
  const int ob   = (rem / 256) * 16;
  const int nb   = (rem % 256) * 16;
  const float* W = (br == 0) ? w0 : (br == 1 ? w1 : w2);
  const float* B = (br == 0) ? b0 : (br == 1 ? b1 : b2);

  v8f acc = {};
#pragma unroll 4
  for (int k0 = 0; k0 < CIN; k0 += 4) {
    v2f a, b;
    a.x = W[(ob + lrow) * CIN + k0 + ka];
    a.y = W[(ob + lrow) * CIN + k0 + ka + 1];
    b.x = x[(k0 + ka) * NTOK + nb + lrow];
    b.y = x[(k0 + ka + 1) * NTOK + nb + lrow];
    acc = wmma_f32_k4(a, b, acc);
  }
  float* out = qkv + (size_t)br * 384 * NTOK;
#pragma unroll
  for (int r = 0; r < 8; ++r) {
    const int row = ob + r + hi * 8;
    out[row * NTOK + nb + lrow] = acc[r] + B[row];
  }
}

// ---------------------------------------------------------------------------
// Kernel 2: one-time f32 -> f16 repack of Q/K/V into matmul-native layouts.
//   Qt[(br,head)*4096 + n][d]  (pre-scaled by DIM_HEAD^-0.5)
//   Kt[(br,head)*4096 + n][d]  (transposed: key-major rows of 16 halfs)
//   V16[(br,head*16+d)*4096 + n]
// plus a 64-half zero page used for the QK K-dim padding.
// ---------------------------------------------------------------------------
__global__ __launch_bounds__(256) void pack_f16_kernel(
    const float* __restrict__ qkv, _Float16* __restrict__ Qt,
    _Float16* __restrict__ Kt, _Float16* __restrict__ V16,
    _Float16* __restrict__ zbuf) {
  const int idx = blockIdx.x * 256 + threadIdx.x;  // 3*384*4096 elements
  const int n = idx & (NTOK - 1);
  const int t = idx >> 12;  // br*384 + row
  const int br = t / 384;
  const int c  = t % 384;
  const float v = qkv[(size_t)t * NTOK + n];
  const int sec  = c >> 7;   // 0=Q, 1=K, 2=V
  const int cc   = c & 127;  // head*16 + d
  const int head = cc >> 4;
  const int d    = cc & 15;
  const size_t hn = ((size_t)(br * 8 + head) * NTOK + n) * 16 + d;
  if (sec == 0)
    Qt[hn] = (_Float16)(v * 0.25f);
  else if (sec == 1)
    Kt[hn] = (_Float16)v;
  else
    V16[((size_t)br * 128 + cc) * NTOK + n] = (_Float16)v;
  if (idx < 64) zbuf[idx] = (_Float16)0.0f;
}

// ---------------------------------------------------------------------------
// Kernel 3: streaming (flash) attention, one (branch, head, q-tile) per wave
// tiles = 3 * 8 * 256 = 6144.  f16 WMMA pipe, 32 keys per iteration:
//   S  = Q^T K : 2x 16x16x32 f16 wmma (d=16 zero-padded via zero-page B rows)
//   O += P V^T : 1x 16x16x32 f16 wmma (full K utilization)
// ---------------------------------------------------------------------------
__global__ __launch_bounds__(256) void attn_kernel(
    const _Float16* __restrict__ Qt, const _Float16* __restrict__ Kt,
    const _Float16* __restrict__ V16, const _Float16* __restrict__ zbuf,
    float* __restrict__ attout) {
  __shared__ _Float16 plds[8 * 16 * 40];  // per-wave 16x32 f16 P tile, stride 40
  const int wave = threadIdx.x >> 5;
  const int lane = threadIdx.x & 31;
  const int lrow = lane & 15;
  const int hi   = lane >> 4;
  const int tile = blockIdx.x * 8 + wave;
  const int br   = tile / (8 * 256);
  const int rem  = tile % (8 * 256);
  const int head = rem / 256;
  const int qb   = (rem % 256) * 16;
  const _Float16* Qh = Qt + (size_t)(br * 8 + head) * NTOK * 16;
  const _Float16* Kh = Kt + (size_t)(br * 8 + head) * NTOK * 16;
  const _Float16* Vh = V16 + (size_t)(br * 128 + head * 16) * NTOK;
  _Float16* ptile = plds + wave * (16 * 40);

  const int dbase = hi * 8;  // D-layout: lanes<16 hold rows 0..7, hi rows 8..15

  // Q^T tile, A-layout: lanes<16 hold K=0..15 (d), hi lanes K=16..31 zero: one aligned 32B load
  v16h qa = {};
  {
    if (!hi) qa = *(const v16h*)(Qh + (size_t)(qb + lrow) * 16);
  }

  float mrow[8], lsum[8];
#pragma unroll
  for (int r = 0; r < 8; ++r) { mrow[r] = -3.0e38f; lsum[r] = 0.0f; }
  v8f o = {};

  for (int kt = 0; kt < 128; ++kt) {
    const int kb = kt * 32;
    const int kbn = (kb + 32) & (NTOK - 1);  // wrap-clamped prefetch target
    __builtin_prefetch(Kh + (size_t)(kbn + lrow) * 16, 0, 1);
    __builtin_prefetch(Vh + lrow * NTOK + kbn, 0, 1);

    // S tiles: B lanes 0-15 carry K-rows 0..15 (= d); hi lanes read the
    // zero page (pointer select, not per-element selects).
    const _Float16* kp0 = hi ? zbuf : Kh + (size_t)(kb + lrow) * 16;
    const _Float16* kp1 = hi ? zbuf : Kh + (size_t)(kb + 16 + lrow) * 16;
    const v16h kb0 = *(const v16h*)kp0;
    const v16h kb1 = *(const v16h*)kp1;
    v8f s0 = {}, s1 = {};
    s0 = wmma_f16_k32(qa, kb0, s0);
    s1 = wmma_f16_k32(qa, kb1, s1);

    // online softmax across the 32-key strip (rows span 16-lane halves)
#pragma unroll
    for (int r = 0; r < 8; ++r) {
      const float sv0 = s0[r];
      const float sv1 = s1[r];
      float t = fmaxf(sv0, sv1);
      t = fmaxf(t, __shfl_xor(t, 1, 32));
      t = fmaxf(t, __shfl_xor(t, 2, 32));
      t = fmaxf(t, __shfl_xor(t, 4, 32));
      t = fmaxf(t, __shfl_xor(t, 8, 32));
      const float mnew  = fmaxf(mrow[r], t);
      const float alpha = __expf(mrow[r] - mnew);
      mrow[r] = mnew;
      const float p0 = __expf(sv0 - mnew);
      const float p1 = __expf(sv1 - mnew);
      float ps = p0 + p1;
      ps += __shfl_xor(ps, 1, 32);
      ps += __shfl_xor(ps, 2, 32);
      ps += __shfl_xor(ps, 4, 32);
      ps += __shfl_xor(ps, 8, 32);
      lsum[r] = lsum[r] * alpha + ps;
      o[r] *= alpha;
      ptile[(r + dbase) * 40 + lrow]      = (_Float16)p0;  // D-layout rows
      ptile[(r + dbase) * 40 + 16 + lrow] = (_Float16)p1;
    }
    __syncthreads();

    // O += P(16x32 A-layout: two aligned ds_load_b128) * V^T(32x16 B-layout)
    {
      const v8h plo = *(const v8h*)(ptile + lrow * 40 + hi * 16);
      const v8h phi = *(const v8h*)(ptile + lrow * 40 + hi * 16 + 8);
      v16h pa;
#pragma unroll
      for (int e = 0; e < 8; ++e) { pa[e] = plo[e]; pa[8 + e] = phi[e]; }
      const v16h vb = *(const v16h*)(Vh + lrow * NTOK + kb + hi * 16);
      o = wmma_f16_k32(pa, vb, o);
    }
    __syncthreads();
  }

  // normalize and store: attout[(head*16 + d) * N + n], d = lrow, n = qb+row
  float* out = attout + (size_t)br * 128 * NTOK + head * 16 * NTOK;
#pragma unroll
  for (int r = 0; r < 8; ++r) {
    out[lrow * NTOK + qb + r + hi * 8] = o[r] * (1.0f / lsum[r]);
  }
}

// ---------------------------------------------------------------------------
// Kernel 4: out = gamma * (sum_br Wout_br * attout_br + sum_br b_br) + x
// tiles = 8 * 256 = 2048  (fp32 wmma)
// ---------------------------------------------------------------------------
__global__ __launch_bounds__(256) void outproj_kernel(
    const float* __restrict__ x, const float* __restrict__ attout,
    const float* __restrict__ w0, const float* __restrict__ bo0,
    const float* __restrict__ w1, const float* __restrict__ bo1,
    const float* __restrict__ w2, const float* __restrict__ bo2,
    const float* __restrict__ gamma, float* __restrict__ out) {
  const int wave = threadIdx.x >> 5;
  const int lane = threadIdx.x & 31;
  const int lrow = lane & 15;
  const int hi   = lane >> 4;
  const int ka   = hi * 2;
  const int tile = blockIdx.x * 8 + wave;
  const int ob   = (tile / 256) * 16;
  const int nb   = (tile % 256) * 16;

  v8f acc = {};
  for (int br = 0; br < 3; ++br) {
    const float* W = (br == 0) ? w0 : (br == 1 ? w1 : w2);
    const float* A = attout + (size_t)br * 128 * NTOK;
#pragma unroll 4
    for (int k0 = 0; k0 < 128; k0 += 4) {
      v2f a, b;
      a.x = W[(ob + lrow) * 128 + k0 + ka];
      a.y = W[(ob + lrow) * 128 + k0 + ka + 1];
      b.x = A[(k0 + ka) * NTOK + nb + lrow];
      b.y = A[(k0 + ka + 1) * NTOK + nb + lrow];
      acc = wmma_f32_k4(a, b, acc);
    }
  }
  const float g = gamma[0];
#pragma unroll
  for (int r = 0; r < 8; ++r) {
    const int row = ob + r + hi * 8;
    const float bsum = bo0[row] + bo1[row] + bo2[row];
    out[row * NTOK + nb + lrow] =
        g * (acc[r] + bsum) + x[row * NTOK + nb + lrow];
  }
}

// ---------------------------------------------------------------------------
extern "C" void kernel_launch(void* const* d_in, const int* in_sizes, int n_in,
                              void* d_out, int out_size, void* d_ws,
                              size_t ws_size, hipStream_t stream) {
  const float* x       = (const float*)d_in[0];
  const float* w_qkv_d = (const float*)d_in[1];
  const float* b_qkv_d = (const float*)d_in[2];
  const float* w_out_d = (const float*)d_in[3];
  const float* b_out_d = (const float*)d_in[4];
  const float* w_qkv_h = (const float*)d_in[5];
  const float* b_qkv_h = (const float*)d_in[6];
  const float* w_out_h = (const float*)d_in[7];
  const float* b_out_h = (const float*)d_in[8];
  const float* w_qkv_w = (const float*)d_in[9];
  const float* b_qkv_w = (const float*)d_in[10];
  const float* w_out_w = (const float*)d_in[11];
  const float* b_out_w = (const float*)d_in[12];
  const float* gamma   = (const float*)d_in[13];
  float* out = (float*)d_out;

  // workspace layout (all offsets 32B aligned):
  float* qkv    = (float*)d_ws;                    // 3*384*4096 f32  (18.9 MB)
  float* attout = qkv + (size_t)3 * 384 * NTOK;    // 3*128*4096 f32  ( 6.3 MB)
  _Float16* Qt  = (_Float16*)(attout + (size_t)3 * 128 * NTOK);  // 3 MB
  _Float16* Kt  = Qt + (size_t)3 * 128 * NTOK;                   // 3 MB
  _Float16* V16 = Kt + (size_t)3 * 128 * NTOK;                   // 3 MB
  _Float16* zbuf = V16 + (size_t)3 * 128 * NTOK;                 // 128 B

  // 1) QKV projections (fp32 wmma): 18432 tiles / 8 waves
  qkv_proj_kernel<<<2304, 256, 0, stream>>>(x, w_qkv_d, b_qkv_d, w_qkv_h,
                                            b_qkv_h, w_qkv_w, b_qkv_w, qkv);
  // 2) repack Q/K/V to f16 matmul-native layouts: 3*384*4096 / 256
  pack_f16_kernel<<<18432, 256, 0, stream>>>(qkv, Qt, Kt, V16, zbuf);
  // 3) flash attention on the f16 WMMA pipe: 6144 tiles / 8 waves
  attn_kernel<<<768, 256, 0, stream>>>(Qt, Kt, V16, zbuf, attout);
  // 4) output projection + gamma + residual (fp32 wmma): 2048 tiles / 8 waves
  outproj_kernel<<<256, 256, 0, stream>>>(x, attout, w_out_d, b_out_d, w_out_h,
                                          b_out_h, w_out_w, b_out_w, gamma,
                                          out);
}
